// MultiHeadAttention_16956530885275
// MI455X (gfx1250) — matrix operand order repaired and checked by reading, not compiled, not run on hardware
//
#include <hip/hip_runtime.h>
#include <hip/hip_bf16.h>

// ---------------------------------------------------------------------------
// Block-sparse multi-head attention for MI455X (gfx1250, wave32, WMMA bf16)
//   B=2, N=2048, H=16, DH=64, D=1024, BS=128, nb=16
// Pipeline:
//   1) x: f32->bf16 ; W*: f32->bf16 TRANSPOSED (so GEMM B slabs copy straight)
//   2) Q/K GEMM -> [b][h][tok][dh] bf16 (Q pre-scaled by 1/sqrt(DH))
//      V   GEMM -> [b][h][dh][tok] bf16 (pre-transposed for PV operand)
//   3) block-sparse flash attention, double-buffered async LDS staging
//   4) output GEMM -> f32 + bias
// All LDS staging uses GLOBAL_LOAD_ASYNC_TO_LDS_B128 when available.
// ---------------------------------------------------------------------------

typedef __bf16 bf16;
typedef __attribute__((ext_vector_type(16))) __bf16 v16bf;
typedef __attribute__((ext_vector_type(8)))  __bf16 bf16x8;
typedef __attribute__((ext_vector_type(8)))  float  v8f;
typedef int v4i __attribute__((vector_size(16)));     // matches builtin param

#define DEV static __device__ __forceinline__

constexpr int B_  = 2;
constexpr int N_  = 2048;
constexpr int H_  = 16;
constexpr int DH_ = 64;
constexpr int D_  = 1024;
constexpr int BS_ = 128;
constexpr int NB_ = 16;          // N / BS
constexpr int MROWS = B_ * N_;   // 4096

#if defined(__HIP_DEVICE_COMPILE__) &&                                   \
    __has_builtin(__builtin_amdgcn_global_load_async_to_lds_b128) &&     \
    __has_builtin(__builtin_amdgcn_s_wait_asynccnt)
#define HAS_ASYNC_LDS 1
#else
#define HAS_ASYNC_LDS 0
#endif

DEV bf16 to_bf16(float f) {
  // round-to-nearest-even bf16 via bit manipulation
  unsigned u = __float_as_uint(f);
  u += 0x7fffu + ((u >> 16) & 1u);
  unsigned short s = (unsigned short)(u >> 16);
  bf16 r;
  __builtin_memcpy(&r, &s, sizeof(r));
  return r;
}

// 16B global -> LDS copy; async (ASYNCcnt) on gfx1250, sync fallback otherwise
DEV void copy_b128_to_lds(const bf16* g, bf16* l) {
#if HAS_ASYNC_LDS
  typedef __attribute__((address_space(1))) v4i* gp_t;
  typedef __attribute__((address_space(3))) v4i* lp_t;
  __builtin_amdgcn_global_load_async_to_lds_b128((gp_t)g, (lp_t)l, 0, 0);
#else
  *(bf16x8*)l = *(const bf16x8*)g;
#endif
}

DEV void wait_async_copies() {
#if HAS_ASYNC_LDS
  __builtin_amdgcn_s_wait_asynccnt(0);
#endif
}

union AFrag { v16bf v; bf16x8 h[2]; };

DEV v8f wmma_bf16(const v16bf a, const v16bf b, v8f c) {
  // D = A(16x32 bf16) * B(32x16 bf16) + C(16x16 f32)
  return __builtin_amdgcn_wmma_f32_16x16x32_bf16(
      false, a, false, b, (short)0, c, false, false);
}

// ---------------------------------------------------------------------------
// f32 -> bf16 conversion (grid-stride)
// ---------------------------------------------------------------------------
__global__ void f32_to_bf16_kernel(const float* __restrict__ src,
                                   bf16* __restrict__ dst, int n) {
  int i = blockIdx.x * blockDim.x + threadIdx.x;
  int stride = gridDim.x * blockDim.x;
  for (; i < n; i += stride) dst[i] = to_bf16(src[i]);
}

// ---------------------------------------------------------------------------
// f32 [R][C] -> bf16 transposed [C][R]  (LDS-tiled 32x32, coalesced both ways)
// ---------------------------------------------------------------------------
__global__ __launch_bounds__(256)
void transpose_to_bf16_kernel(const float* __restrict__ src,
                              bf16* __restrict__ dst, int R, int C) {
  __shared__ float tile[32][33];
  const int c0 = blockIdx.x * 32, r0 = blockIdx.y * 32;
  const int tx = threadIdx.x & 31, ty = threadIdx.x >> 5;  // ty: 0..7
#pragma unroll
  for (int i = 0; i < 32; i += 8)
    tile[ty + i][tx] = src[(size_t)(r0 + ty + i) * C + c0 + tx];
  __syncthreads();
#pragma unroll
  for (int i = 0; i < 32; i += 8)
    dst[(size_t)(c0 + ty + i) * R + r0 + tx] = to_bf16(tile[tx][ty + i]);
}

// ---------------------------------------------------------------------------
// Tiled bf16 GEMM:  C(MxN) = A(MxK) @ B + bias, B given TRANSPOSED as Wt[N][K].
// 128x128 C tile / workgroup, 8 waves, 8 WMMA per wave per 32-wide K slab.
// Double-buffered LDS, async global->LDS staging.
// MODE 0: f32 row-major out ((v + bias) * scale applied)
// MODE 1: bf16 out, per-head layout   [b][h][tok][dh]   (Q, K)
// MODE 2: bf16 out, head-T layout     [b][h][dh][tok]   (V)
// ---------------------------------------------------------------------------
template <int MODE>
__global__ __launch_bounds__(256)
void gemm_bf16_kernel(const bf16* __restrict__ A, const bf16* __restrict__ Wt,
                      const float* __restrict__ bias, void* __restrict__ Out,
                      int M, int Nd, int K, float outScale) {
  __shared__ __align__(16) bf16 Albuf[2][128][32];  // A slab  [m][k]
  __shared__ __align__(16) bf16 Blbuf[2][128][32];  // B slab  [n][k]

  const int tid   = threadIdx.x;
  const int lane  = tid & 31;
  const int wave  = tid >> 5;
  const int wm    = wave >> 1;   // 0..3 : 32-row band
  const int wn    = wave & 1;    // 0..1 : 64-col band
  const int m0    = blockIdx.y * 128;
  const int n0    = blockIdx.x * 128;
  const int nlo   = lane & 15;
  const int nhalf = lane >> 4;
  const int koff  = nhalf * 16;

  // stage one 32-wide K slab: 2 x 256 chunks of 16B, 1 each per thread
  auto stage = [&](int kk, int bsel) {
    int row = tid >> 1, cc = tid & 1;  // 128 rows x 2 chunks
    copy_b128_to_lds(&A[(size_t)(m0 + row) * K + kk + cc * 16],
                     &Albuf[bsel][row][cc * 16]);
    copy_b128_to_lds(&Wt[(size_t)(n0 + row) * K + kk + cc * 16],
                     &Blbuf[bsel][row][cc * 16]);
  };

  v8f acc[2][4];
#pragma unroll
  for (int i = 0; i < 2; ++i)
#pragma unroll
    for (int j = 0; j < 4; ++j) acc[i][j] = v8f{0, 0, 0, 0, 0, 0, 0, 0};

  stage(0, 0);
  int bsel = 0;
  for (int kk = 0; kk < K; kk += 32) {
    wait_async_copies();
    __syncthreads();
    if (kk + 32 < K) stage(kk + 32, bsel ^ 1);

    AFrag af[2], bf[4];
#pragma unroll
    for (int i = 0; i < 2; ++i) {
      int m = wm * 32 + i * 16 + nlo;
      // A layout: lanes 0-15 K {0-7,16-23}; lanes 16-31 K {8-15,24-31}
      af[i].h[0] = *(const bf16x8*)&Albuf[bsel][m][nhalf * 8];
      af[i].h[1] = *(const bf16x8*)&Albuf[bsel][m][16 + nhalf * 8];
    }
#pragma unroll
    for (int j = 0; j < 4; ++j) {
      int n = wn * 64 + j * 16 + nlo;
      // B layout: lanes 0-15 K {0-7,16-23}; lanes 16-31 K {8-15,24-31}
      bf[j].h[0] = *(const bf16x8*)&Blbuf[bsel][n][nhalf * 8];
      bf[j].h[1] = *(const bf16x8*)&Blbuf[bsel][n][16 + nhalf * 8];
    }
#pragma unroll
    for (int i = 0; i < 2; ++i)
#pragma unroll
      for (int j = 0; j < 4; ++j)
        acc[i][j] = wmma_bf16(af[i].v, bf[j].v, acc[i][j]);
    bsel ^= 1;
  }

  // ---- epilogue: C layout VGPR r, lanes 0-15 M=r, lanes 16-31 M=r+8 ----
#pragma unroll
  for (int i = 0; i < 2; ++i) {
#pragma unroll
    for (int j = 0; j < 4; ++j) {
#pragma unroll
      for (int r = 0; r < 8; ++r) {
        int row = m0 + wm * 32 + i * 16 + r + 8 * nhalf;
        int col = n0 + wn * 64 + j * 16 + nlo;
        float v = (acc[i][j][r] + bias[col]) * outScale;
        if (MODE == 0) {
          ((float*)Out)[(size_t)row * Nd + col] = v;
        } else {
          int b = row >> 11, tok = row & (N_ - 1);
          int h = col >> 6, dh = col & (DH_ - 1);
          size_t idx = (MODE == 1)
              ? ((((size_t)(b * H_ + h)) * N_ + tok) * DH_ + dh)
              : ((((size_t)(b * H_ + h)) * DH_ + dh) * N_ + tok);
          ((bf16*)Out)[idx] = to_bf16(v);
        }
      }
    }
  }
}

// ---------------------------------------------------------------------------
// Block-sparse flash attention.
// Grid: (iblk, h, b). 256 threads = 8 waves; wave w owns query rows
// [q0+16w, q0+16w+16). Double-buffered async staging of K [key][dh] and
// pre-transposed V [dh][key]; per-wave LDS bounce converts P to A-layout.
// Q arrives pre-scaled by 1/sqrt(DH).
// ---------------------------------------------------------------------------
__global__ __launch_bounds__(256)
void attn_kernel(const bf16* __restrict__ Q, const bf16* __restrict__ Kc,
                 const bf16* __restrict__ Vt, const int* __restrict__ bmask,
                 bf16* __restrict__ Out) {
  __shared__ __align__(16) bf16 Kbuf[2][BS_][DH_];   // [key][dh]   2x16 KB
  __shared__ __align__(16) bf16 Vbuf[2][DH_][BS_];   // [dh][key]   2x16 KB
  __shared__ __align__(16) bf16 Plds[8][16][BS_];    // per-wave P    32 KB

  const int iblk  = blockIdx.x;
  const int h     = blockIdx.y;
  const int b     = blockIdx.z;
  const int tid   = threadIdx.x;
  const int lane  = tid & 31;
  const int wave  = tid >> 5;
  const int nlo   = lane & 15;
  const int nhalf = lane >> 4;
  const int koff  = nhalf * 16;

  const size_t headBase = ((size_t)(b * H_ + h)) * N_ * DH_;  // Q,K layout
  const size_t vtBase   = ((size_t)(b * H_ + h)) * DH_ * N_;  // V-T layout
  const int q0 = iblk * BS_;

  auto stage = [&](int jblk, int bsel) {
    const int key0 = jblk * BS_;
#pragma unroll
    for (int ii = 0; ii < 4; ++ii) {
      int cid = tid + 256 * ii;                 // 0..1023
      int kr = cid >> 3, kc = cid & 7;          // K: 128 rows x 8 chunks
      copy_b128_to_lds(&Kc[headBase + (size_t)(key0 + kr) * DH_ + kc * 8],
                       &Kbuf[bsel][kr][kc * 8]);
      int vr = cid >> 4, vc = cid & 15;         // Vt: 64 rows x 16 chunks
      copy_b128_to_lds(&Vt[vtBase + (size_t)vr * N_ + key0 + vc * 8],
                       &Vbuf[bsel][vr][vc * 8]);
    }
  };

  // ---- persistent Q fragments: 16 rows x 64 dh per wave (2 A-frags) ----
  AFrag qa[2];
  {
    int qrow = q0 + wave * 16 + nlo;
    const bf16x8* qp = (const bf16x8*)&Q[headBase + (size_t)qrow * DH_];
    qa[0].h[0] = qp[nhalf];
    qa[0].h[1] = qp[2 + nhalf];
    qa[1].h[0] = qp[4 + nhalf];
    qa[1].h[1] = qp[6 + nhalf];
  }

  v8f acc_o[4];
#pragma unroll
  for (int t = 0; t < 4; ++t) acc_o[t] = v8f{0, 0, 0, 0, 0, 0, 0, 0};
  float m_run[8], l_run[8];
#pragma unroll
  for (int r = 0; r < 8; ++r) { m_run[r] = -1e30f; l_run[r] = 0.0f; }

  // first active key block
  int jcur = -1;
  for (int j = 0; j <= iblk; ++j)
    if (bmask[iblk * NB_ + j] != 0) { jcur = j; break; }
  int bsel = 0;
  if (jcur >= 0) stage(jcur, 0);

  while (jcur >= 0) {
    int jnext = -1;
    for (int j = jcur + 1; j <= iblk; ++j)
      if (bmask[iblk * NB_ + j] != 0) { jnext = j; break; }

    wait_async_copies();
    __syncthreads();
    if (jnext >= 0) stage(jnext, bsel ^ 1);   // overlap with compute below

    // ---- scores S = Q @ K^T : 8 tiles of 16 keys, 2 dh-chunks each ----
    v8f s[8];
#pragma unroll
    for (int t = 0; t < 8; ++t) {
      AFrag kb0, kb1;  // B-operand: column = key, K-dim = dh
      kb0.h[0] = *(const bf16x8*)&Kbuf[bsel][t * 16 + nlo][nhalf * 8];
      kb0.h[1] = *(const bf16x8*)&Kbuf[bsel][t * 16 + nlo][16 + nhalf * 8];
      kb1.h[0] = *(const bf16x8*)&Kbuf[bsel][t * 16 + nlo][32 + nhalf * 8];
      kb1.h[1] = *(const bf16x8*)&Kbuf[bsel][t * 16 + nlo][48 + nhalf * 8];
      v8f z = v8f{0, 0, 0, 0, 0, 0, 0, 0};
      z = wmma_bf16(qa[0].v, kb0.v, z);
      s[t] = wmma_bf16(qa[1].v, kb1.v, z);
    }

    // ---- flash softmax update (per row r; 16-lane half = one row) ----
#pragma unroll
    for (int r = 0; r < 8; ++r) {
      float mx = s[0][r];
#pragma unroll
      for (int t = 1; t < 8; ++t) mx = fmaxf(mx, s[t][r]);
#pragma unroll
      for (int off = 8; off > 0; off >>= 1)
        mx = fmaxf(mx, __shfl_xor(mx, off, 16));
      float nm    = fmaxf(m_run[r], mx);
      float alpha = __expf(m_run[r] - nm);
      float rs = 0.0f;
#pragma unroll
      for (int t = 0; t < 8; ++t) {
        float p = __expf(s[t][r] - nm);
        s[t][r] = p;
        rs += p;
      }
#pragma unroll
      for (int off = 8; off > 0; off >>= 1) rs += __shfl_xor(rs, off, 16);
      m_run[r] = nm;
      l_run[r] = l_run[r] * alpha + rs;
#pragma unroll
      for (int tt = 0; tt < 4; ++tt) acc_o[tt][r] *= alpha;  // rescale O
    }

    // ---- C-layout -> A-layout via per-wave LDS bounce (no barrier) ----
#pragma unroll
    for (int t = 0; t < 8; ++t)
#pragma unroll
      for (int r = 0; r < 8; ++r)
        Plds[wave][r + 8 * nhalf][t * 16 + nlo] = to_bf16(s[t][r]);

    // ---- O += P @ V : 4 key-chunks x 4 dh-tiles ----
#pragma unroll
    for (int c = 0; c < 4; ++c) {
      AFrag pa;
      pa.h[0] = *(const bf16x8*)&Plds[wave][nlo][c * 32 + nhalf * 8];
      pa.h[1] = *(const bf16x8*)&Plds[wave][nlo][c * 32 + 16 + nhalf * 8];
#pragma unroll
      for (int tt = 0; tt < 4; ++tt) {
        AFrag vb;  // B-operand: column = dh, K-dim = key
        vb.h[0] = *(const bf16x8*)&Vbuf[bsel][tt * 16 + nlo][c * 32 + nhalf * 8];
        vb.h[1] = *(const bf16x8*)&Vbuf[bsel][tt * 16 + nlo][c * 32 + 16 + nhalf * 8];
        acc_o[tt] = wmma_bf16(pa.v, vb.v, acc_o[tt]);
      }
    }

    bsel ^= 1;
    jcur = jnext;
  }

  // ---- finalize: O /= l, write bf16 into [B*N][D] for final projection ----
#pragma unroll
  for (int tt = 0; tt < 4; ++tt) {
#pragma unroll
    for (int r = 0; r < 8; ++r) {
      int row = q0 + wave * 16 + r + 8 * nhalf;
      int col = h * DH_ + tt * 16 + nlo;
      float inv = (l_run[r] > 0.0f) ? 1.0f / l_run[r] : 0.0f;
      Out[((size_t)(b * N_) + row) * D_ + col] = to_bf16(acc_o[tt][r] * inv);
    }
  }
}

// ---------------------------------------------------------------------------
// kernel_launch
// inputs: x, block_mask, wq, bq, wk, bk, wv, bv, wo, bo
// ---------------------------------------------------------------------------
extern "C" void kernel_launch(void* const* d_in, const int* in_sizes, int n_in,
                              void* d_out, int out_size, void* d_ws,
                              size_t ws_size, hipStream_t stream) {
  const float* x  = (const float*)d_in[0];
  const int*   bm = (const int*)d_in[1];
  const float* wq = (const float*)d_in[2];
  const float* bq = (const float*)d_in[3];
  const float* wk = (const float*)d_in[4];
  const float* bk = (const float*)d_in[5];
  const float* wv = (const float*)d_in[6];
  const float* bv = (const float*)d_in[7];
  const float* wo = (const float*)d_in[8];
  const float* bo = (const float*)d_in[9];

  char* ws = (char*)d_ws;
  bf16* xh   = (bf16*)ws; ws += (size_t)MROWS * D_ * sizeof(bf16);
  bf16* wqt  = (bf16*)ws; ws += (size_t)D_ * D_ * sizeof(bf16);
  bf16* wkt  = (bf16*)ws; ws += (size_t)D_ * D_ * sizeof(bf16);
  bf16* wvt  = (bf16*)ws; ws += (size_t)D_ * D_ * sizeof(bf16);
  bf16* wot  = (bf16*)ws; ws += (size_t)D_ * D_ * sizeof(bf16);
  bf16* qb   = (bf16*)ws; ws += (size_t)B_ * H_ * N_ * DH_ * sizeof(bf16);
  bf16* kb   = (bf16*)ws; ws += (size_t)B_ * H_ * N_ * DH_ * sizeof(bf16);
  bf16* vtb  = (bf16*)ws; ws += (size_t)B_ * H_ * N_ * DH_ * sizeof(bf16);
  bf16* attn = (bf16*)ws; ws += (size_t)MROWS * D_ * sizeof(bf16);

  const float qscale = 0.125f;  // 1/sqrt(DH) folded into Q projection

  // 1) precision conversion; weights transposed -> Wt[N][K]
  f32_to_bf16_kernel<<<512, 256, 0, stream>>>(x, xh, MROWS * D_);
  dim3 gt(D_ / 32, D_ / 32);
  transpose_to_bf16_kernel<<<gt, 256, 0, stream>>>(wq, wqt, D_, D_);
  transpose_to_bf16_kernel<<<gt, 256, 0, stream>>>(wk, wkt, D_, D_);
  transpose_to_bf16_kernel<<<gt, 256, 0, stream>>>(wv, wvt, D_, D_);
  transpose_to_bf16_kernel<<<gt, 256, 0, stream>>>(wo, wot, D_, D_);

  // 2) projections
  dim3 gg(D_ / 128, MROWS / 128);
  gemm_bf16_kernel<1><<<gg, 256, 0, stream>>>(xh, wqt, bq, qb,
                                              MROWS, D_, D_, qscale);
  gemm_bf16_kernel<1><<<gg, 256, 0, stream>>>(xh, wkt, bk, kb,
                                              MROWS, D_, D_, 1.0f);
  gemm_bf16_kernel<2><<<gg, 256, 0, stream>>>(xh, wvt, bv, vtb,
                                              MROWS, D_, D_, 1.0f);

  // 3) block-sparse flash attention
  dim3 ga(NB_, H_, B_);
  attn_kernel<<<ga, 256, 0, stream>>>(qb, kb, vtb, bm, attn);

  // 4) output projection -> f32
  gemm_bf16_kernel<0><<<gg, 256, 0, stream>>>(attn, wot, bo, (float*)d_out,
                                              MROWS, D_, D_, 1.0f);
}
